// CausalAttention_55379308314860
// MI455X (gfx1250) — matrix operand order repaired and checked by reading, not compiled, not run on hardware
//
#include <hip/hip_runtime.h>

typedef _Float16 v16h __attribute__((ext_vector_type(16)));
typedef _Float16 v8h  __attribute__((ext_vector_type(8)));
typedef float    v8f  __attribute__((ext_vector_type(8)));
typedef float    v4f  __attribute__((ext_vector_type(4)));

constexpr int B_ = 4;
constexpr int S_ = 4096;
constexpr int DE = 512;
constexpr int DK = 64;   // == d_v

// ---------------------------------------------------------------------------
// gfx1250 async copy: 16B global -> LDS, tracked by ASYNCcnt.
// VDST VGPR carries the LDS byte offset, VADDR the 64-bit global address.
// ---------------------------------------------------------------------------
__device__ __forceinline__ void async_cp16(const _Float16* g, _Float16* l)
{
    unsigned long long ga = (unsigned long long)(uintptr_t)g;
    unsigned int       la = (unsigned int)(uintptr_t)l;
    asm volatile("global_load_async_to_lds_b128 %0, %1, off"
                 :: "v"(la), "v"(ga) : "memory");
}
__device__ __forceinline__ void async_wait0()
{
    asm volatile("s_wait_asynccnt 0x0" ::: "memory");
}

// ---------------------------------------------------------------------------
// Kernel 1: QKV projection.  q,k stored row-major f16 [B*S][64]; v stored
// transposed f16 [B][64][S].  Each wave computes a 16x64 tile via
// v_wmma_f32_16x16x32_f16.  Store epilogue is branchless via (rs,cs,ex).
// ---------------------------------------------------------------------------
__global__ __launch_bounds__(256)
void qkv_proj_kernel(const float* __restrict__ x,
                     const float* __restrict__ Wq,
                     const float* __restrict__ Wk,
                     const float* __restrict__ Wv,
                     _Float16* __restrict__ qo,
                     _Float16* __restrict__ ko,
                     _Float16* __restrict__ vto)
{
    __shared__ _Float16 wt[DK][256 + 8];   // transposed W, row stride 528B (16B aligned)

    const int mat = blockIdx.y;            // 0=Q 1=K 2=V
    const float* W = (mat == 0) ? Wq : (mat == 1) ? Wk : Wv;
    const int tid  = threadIdx.x;
    const int wave = tid >> 5;
    const int lane = tid & 31;
    const int m    = lane & 15;
    const int hi   = lane >> 4;
    const int row0 = blockIdx.x * 128 + wave * 16;

    v8f acc[4] = { {}, {}, {}, {} };

    const float* xrow = x + (size_t)(row0 + m) * DE;

    for (int kh = 0; kh < DE; kh += 256) {
        __syncthreads();
        for (int i = tid; i < 256 * DK; i += 256) {
            const int kk = i >> 6;
            const int n  = i & 63;
            wt[n][kk] = (_Float16)W[(size_t)(kh + kk) * DK + n];
        }
        __syncthreads();

        for (int kb = 0; kb < 256; kb += 32) {
            // ---- A tile (16x32 of x, f32 -> f16), ISA A-layout ----
            const int koff = kh + kb + hi * 8;
            v4f x0 = *(const v4f*)(xrow + koff);
            v4f x1 = *(const v4f*)(xrow + koff + 4);
            v4f x2 = *(const v4f*)(xrow + koff + 16);
            v4f x3 = *(const v4f*)(xrow + koff + 20);
            v16h a;
            #pragma unroll
            for (int i = 0; i < 4; ++i) {
                a[i]      = (_Float16)x0[i];
                a[4 + i]  = (_Float16)x1[i];
                a[8 + i]  = (_Float16)x2[i];
                a[12 + i] = (_Float16)x3[i];
            }
            // ---- load all 4 B tiles first (batched dscnt waits), then WMMAs ----
            v16h bmv[4];
            #pragma unroll
            for (int nt = 0; nt < 4; ++nt) {
                const _Float16* wp = &wt[nt * 16 + m][kb + hi * 8];
                v8h blo = *(const v8h*)wp;
                v8h bhi = *(const v8h*)(wp + 16);
                #pragma unroll
                for (int i = 0; i < 8; ++i) { bmv[nt][i] = blo[i]; bmv[nt][8 + i] = bhi[i]; }
            }
            #pragma unroll
            for (int nt = 0; nt < 4; ++nt) {
                acc[nt] = __builtin_amdgcn_wmma_f32_16x16x32_f16(
                    false, a, false, bmv[nt], (short)0, acc[nt], false, false);
            }
        }
    }

    // ---- branchless store epilogue ----
    // mat 0/1 : addr = row*DK + col               (rs=DK, cs=1,  ex=0)
    // mat 2   : addr = (b*DK+col)*S + s
    //         = row*1 + col*S + (row>>12)*(DK-1)*S (rs=1,  cs=S,  ex=(DK-1)*S)
    _Float16* dst;
    size_t rs, cs, ex;
    if (mat == 0)      { dst = qo;  rs = DK; cs = 1;  ex = 0; }
    else if (mat == 1) { dst = ko;  rs = DK; cs = 1;  ex = 0; }
    else               { dst = vto; rs = 1;  cs = S_; ex = (size_t)(DK - 1) * S_; }

    #pragma unroll
    for (int nt = 0; nt < 4; ++nt) {
        #pragma unroll
        for (int r = 0; r < 8; ++r) {
            const size_t row = (size_t)(row0 + r + 8 * hi);
            const size_t col = (size_t)(nt * 16 + m);
            dst[row * rs + col * cs + (row >> 12) * ex] = (_Float16)acc[nt][r];
        }
    }
}

// ---------------------------------------------------------------------------
// Kernel 2: flash-attention.  Block = 4 waves = 64 query rows.  K/V chunks of
// 32 keys are double-buffered in LDS via global_load_async_to_lds_b128
// (ASYNCcnt), shared by all 4 waves.  All matmuls via v_wmma_f32_16x16x32_f16.
// ---------------------------------------------------------------------------
__global__ __launch_bounds__(128)
void flash_attn_kernel(const _Float16* __restrict__ q,
                       const _Float16* __restrict__ k,
                       const _Float16* __restrict__ vt,
                       float* __restrict__ out)
{
    __shared__ _Float16 Kb[2][32][72];      // [buf][key][dk], padded rows (144B)
    __shared__ _Float16 Vb[2][64][40];      // [buf][dv][key], padded rows (80B)
    __shared__ _Float16 pstage[4][16 * 32]; // per-wave P staging (C->A relayout)

    const int tid  = threadIdx.x;
    const int wave = tid >> 5;
    const int lane = tid & 31;
    const int m    = lane & 15;
    const int hi   = lane >> 4;

    const int b   = blockIdx.x >> 6;            // / (S/64)
    const int Q0  = (blockIdx.x & 63) * 64;     // block's first query row
    const int q0  = Q0 + wave * 16;             // wave's query tile

    const _Float16* kbase = k  + (size_t)b * S_ * DK;
    const _Float16* vbase = vt + (size_t)b * DK * S_;

    // ---- Q as two 16x32 A-tiles, kept in registers ----
    const _Float16* qrow = q + ((size_t)b * S_ + q0 + m) * DK;
    v16h qa[2];
    #pragma unroll
    for (int t = 0; t < 2; ++t) {
        const int koff = t * 32 + hi * 8;
        v8h lo  = *(const v8h*)(qrow + koff);
        v8h hi8 = *(const v8h*)(qrow + koff + 16);
        #pragma unroll
        for (int i = 0; i < 8; ++i) { qa[t][i] = lo[i]; qa[t][8 + i] = hi8[i]; }
    }

    v8f acc[4] = { {}, {}, {}, {} };
    float mrow[8], lrow[8];
    #pragma unroll
    for (int r = 0; r < 8; ++r) { mrow[r] = -1e30f; lrow[r] = 0.0f; }

    const int jmax   = q0 + 15;      // per-wave causal bound
    const int jlimit = Q0 + 63;      // block's last needed key

    auto stage_chunk = [&](int j0, int p) {
        #pragma unroll
        for (int it = 0; it < 2; ++it) {
            const int i  = tid + it * 128;
            const int kr = i >> 3, kc = (i & 7) * 8;
            async_cp16(kbase + (size_t)(j0 + kr) * DK + kc, &Kb[p][kr][kc]);
            const int vr = i >> 2, vc = (i & 3) * 8;
            async_cp16(vbase + (size_t)vr * S_ + j0 + vc, &Vb[p][vr][vc]);
        }
    };

    int p = 0;
    stage_chunk(0, 0);
    async_wait0();
    __syncthreads();

    for (int j0 = 0; j0 <= jlimit; j0 += 32) {
        if (j0 + 32 <= jlimit) stage_chunk(j0 + 32, p ^ 1);

        if (j0 <= jmax) {                      // wave-uniform: EXEC stays full
            // ---- scores: S = Q * K^T from LDS ----
            v8f sc[2];
            #pragma unroll
            for (int nt = 0; nt < 2; ++nt) {
                v8f s = {};
                #pragma unroll
                for (int kst = 0; kst < 2; ++kst) {
                    const _Float16* kp = &Kb[p][nt * 16 + m][kst * 32 + hi * 8];
                    v8h lo = *(const v8h*)kp;
                    v8h h8 = *(const v8h*)(kp + 16);
                    v16h bm;
                    #pragma unroll
                    for (int i = 0; i < 8; ++i) { bm[i] = lo[i]; bm[8 + i] = h8[i]; }
                    s = __builtin_amdgcn_wmma_f32_16x16x32_f16(
                        false, qa[kst], false, bm, (short)0, s, false, false);
                }
                sc[nt] = s;
            }

            // ---- scale, causal mask, online softmax ----
            #pragma unroll
            for (int r = 0; r < 8; ++r) {
                const int qi = q0 + r + 8 * hi;
                float s0 = sc[0][r] * 0.125f;
                float s1 = sc[1][r] * 0.125f;
                if (j0 + m > qi)      s0 = -1e30f;
                if (j0 + 16 + m > qi) s1 = -1e30f;
                float t = fmaxf(s0, s1);
                #pragma unroll
                for (int d = 1; d < 16; d <<= 1) t = fmaxf(t, __shfl_xor(t, d, 32));
                const float mn    = fmaxf(mrow[r], t);
                const float rescl = __expf(mrow[r] - mn);
                const float p0    = __expf(s0 - mn);
                const float p1    = __expf(s1 - mn);
                float rs = p0 + p1;
                #pragma unroll
                for (int d = 1; d < 16; d <<= 1) rs += __shfl_xor(rs, d, 32);
                lrow[r] = lrow[r] * rescl + rs;
                mrow[r] = mn;
                #pragma unroll
                for (int t4 = 0; t4 < 4; ++t4) acc[t4][r] *= rescl;
                pstage[wave][(r + 8 * hi) * 32 + m]      = (_Float16)p0;
                pstage[wave][(r + 8 * hi) * 32 + 16 + m] = (_Float16)p1;
            }

            // C-layout -> A-layout relayout through LDS (wave-local, in-order DS)
            __builtin_amdgcn_wave_barrier();
            asm volatile("s_wait_dscnt 0x0" ::: "memory");
            v16h pa;
            {
                const _Float16* pp = &pstage[wave][m * 32 + hi * 8];
                #pragma unroll
                for (int i = 0; i < 8; ++i) { pa[i] = pp[i]; pa[8 + i] = pp[16 + i]; }
            }
            __builtin_amdgcn_wave_barrier();

            // ---- O += P * V from LDS ----
            #pragma unroll
            for (int t4 = 0; t4 < 4; ++t4) {
                const _Float16* vp = &Vb[p][t4 * 16 + m][hi * 8];
                v8h lo = *(const v8h*)vp;
                v8h h8 = *(const v8h*)(vp + 16);
                v16h bm;
                #pragma unroll
                for (int i = 0; i < 8; ++i) { bm[i] = lo[i]; bm[8 + i] = h8[i]; }
                acc[t4] = __builtin_amdgcn_wmma_f32_16x16x32_f16(
                    false, pa, false, bm, (short)0, acc[t4], false, false);
            }
        }

        async_wait0();          // own wave's async copies for next chunk done
        __syncthreads();        // publish LDS to all waves / protect reuse
        p ^= 1;
    }

    // ---- normalize and store fp32 output [B,S,64] ----
    #pragma unroll
    for (int t4 = 0; t4 < 4; ++t4) {
        #pragma unroll
        for (int r = 0; r < 8; ++r) {
            const int row = q0 + r + 8 * hi;
            out[(((size_t)b * S_) + row) * DK + t4 * 16 + m] = acc[t4][r] / lrow[r];
        }
    }
}

// ---------------------------------------------------------------------------
extern "C" void kernel_launch(void* const* d_in, const int* in_sizes, int n_in,
                              void* d_out, int out_size, void* d_ws, size_t ws_size,
                              hipStream_t stream)
{
    const float* x  = (const float*)d_in[0];
    const float* Wq = (const float*)d_in[1];
    const float* Wk = (const float*)d_in[2];
    const float* Wv = (const float*)d_in[3];
    float* out = (float*)d_out;

    const size_t n = (size_t)B_ * S_ * DK;
    _Float16* qws = (_Float16*)d_ws;
    _Float16* kws = qws + n;
    _Float16* vws = kws + n;                 // stored transposed [B][64][S]

    qkv_proj_kernel<<<dim3((B_ * S_) / 128, 3), 256, 0, stream>>>(
        x, Wq, Wk, Wv, qws, kws, vws);

    flash_attn_kernel<<<dim3((B_ * S_) / 64), 128, 0, stream>>>(
        qws, kws, vws, out);
}
